// LanNet_80702435492424
// MI455X (gfx1250) — matrix operand order repaired and checked by reading, not compiled, not run on hardware
//
#include <hip/hip_runtime.h>

#define Bb   32
#define Tt   512
#define Dd   48
#define Hh   2048
#define NG6  12288   // 6*H
#define NG5  10240   // 5*H

typedef __attribute__((ext_vector_type(8)))  unsigned short us8;
typedef __attribute__((ext_vector_type(16))) unsigned short us16;
typedef __attribute__((ext_vector_type(16))) __bf16        v16bf;
typedef __attribute__((ext_vector_type(8)))  float         v8f;

__device__ __forceinline__ unsigned short f2bf_bits(float f) {
  __bf16 h = (__bf16)f;                       // native v_cvt_pk_bf16_f32
  return __builtin_bit_cast(unsigned short, h);
}
__device__ __forceinline__ float bf_bits2f(unsigned short s) {
  unsigned u = ((unsigned)s) << 16;
  return __builtin_bit_cast(float, u);
}
__device__ __forceinline__ float sigmoidf(float x) { return 1.0f / (1.0f + __expf(-x)); }

// Assemble a 16-element bf16 A-fragment from two 16B chunks 32B apart.
// ISA 16-bit A 16x32 layout: elements 0..7 = K {base..base+7}, 8..15 = K {base+16..base+23};
// the lane half supplies the +8 K offset via the caller's addressing.
__device__ __forceinline__ v16bf make_fragA(const unsigned short* p0) {
  us8 a = *(const us8*)p0;
  us8 b = *(const us8*)(p0 + 16);
  us16 r;
#pragma unroll
  for (int i = 0; i < 8; ++i) { r[i] = a[i]; r[i + 8] = b[i]; }
  return __builtin_bit_cast(v16bf, r);
}
__device__ __forceinline__ v16bf load_fragB(const unsigned short* p) {
  return __builtin_bit_cast(v16bf, *(const us16*)p);   // 32B contiguous (pre-packed)
}

__device__ __forceinline__ void grid_barrier(unsigned* cnt, unsigned* gen, unsigned nb) {
  __threadfence();
  __syncthreads();
  if (threadIdx.x == 0) {
    unsigned g = __hip_atomic_load(gen, __ATOMIC_RELAXED, __HIP_MEMORY_SCOPE_AGENT);
    unsigned a = __hip_atomic_fetch_add(cnt, 1u, __ATOMIC_ACQ_REL, __HIP_MEMORY_SCOPE_AGENT);
    if (a == nb - 1u) {
      __hip_atomic_store(cnt, 0u, __ATOMIC_RELAXED, __HIP_MEMORY_SCOPE_AGENT);
      __hip_atomic_store(gen, g + 1u, __ATOMIC_RELEASE, __HIP_MEMORY_SCOPE_AGENT);
    } else {
      while (__hip_atomic_load(gen, __ATOMIC_ACQUIRE, __HIP_MEMORY_SCOPE_AGENT) == g) {
        __builtin_amdgcn_s_sleep(2);
      }
    }
  }
  __syncthreads();
}

// ---------------- prep kernels ----------------

__global__ void k_zero(unsigned* hw, unsigned* bar) {
  unsigned i = blockIdx.x * blockDim.x + threadIdx.x;
  if (i < (2u * Bb * Hh / 2u)) hw[i] = 0u;   // both h buffers (bf16) as u32 words
  if (i < 2u) bar[i] = 0u;
}

// Pack W_state (2048 x 10240 f32, K x N row-major) into wave32 bf16 B-fragment tiles:
// layout [g(5)][j(128)][kk(64)][lane(32)][16]; lane -> col (lane&15), K half (lane>>4).
__global__ void k_pack_ws(const float* __restrict__ W, unsigned short* __restrict__ out) {
  size_t i = (size_t)blockIdx.x * 256 + threadIdx.x;
  if (i >= (size_t)5 * 128 * 64 * 32 * 16) return;
  int e  = (int)(i & 15);
  int l  = (int)((i >> 4) & 31);
  int kk = (int)((i >> 9) & 63);
  int j  = (int)((i >> 15) & 127);
  int g  = (int)(i >> 22);
  int col  = g * Hh + j * 16 + (l & 15);
  int krow = kk * 32 + (l >> 4) * 8 + (e >> 3) * 16 + (e & 7);
  out[i] = f2bf_bits(W[(size_t)krow * NG5 + col]);
}

// Pack W_in (48 x 12288 f32) into B-fragment tiles, K padded 48->64:
// layout [nt(768)][kk(2)][lane(32)][16]
__global__ void k_pack_win(const float* __restrict__ W, unsigned short* __restrict__ out) {
  size_t i = (size_t)blockIdx.x * 256 + threadIdx.x;
  if (i >= (size_t)768 * 2 * 32 * 16) return;
  int e  = (int)(i & 15);
  int l  = (int)((i >> 4) & 31);
  int kk = (int)((i >> 9) & 1);
  int nt = (int)(i >> 10);
  int col  = nt * 16 + (l & 15);
  int krow = kk * 32 + (l >> 4) * 8 + (e >> 3) * 16 + (e & 7);
  float v = (krow < Dd) ? W[(size_t)krow * NG6 + col] : 0.0f;
  out[i] = f2bf_bits(v);
}

// src (16384 x 48 f32) -> bf16 row-major padded to 64 cols
__global__ void k_pack_src(const float* __restrict__ src, unsigned short* __restrict__ out) {
  size_t i = (size_t)blockIdx.x * 256 + threadIdx.x;
  if (i >= (size_t)Bb * Tt * 64) return;
  int k = (int)(i & 63);
  size_t row = i >> 6;
  float v = (k < Dd) ? src[row * Dd + k] : 0.0f;
  out[i] = f2bf_bits(v);
}

// ---------------- GEMM 1: pi = src @ W_in (bf16 WMMA, f32 acc, bf16 out) ----------------

__global__ void __launch_bounds__(256) k_gemm_pi(const unsigned short* __restrict__ srcp,
                                                 const unsigned short* __restrict__ winp,
                                                 unsigned short* __restrict__ pi) {
  const int lane = threadIdx.x & 31;
  const int wave = threadIdx.x >> 5;
  const int tile = blockIdx.x * 8 + wave;        // 786432 tiles total
  const int mt = tile / 768;
  const int nt = tile - mt * 768;
  const int li = lane & 15, lh = lane >> 4;

  v8f acc = {};
#pragma unroll
  for (int kk = 0; kk < 2; ++kk) {
    const unsigned short* a0 = srcp + ((size_t)(mt * 16 + li) * 64 + kk * 32 + lh * 8);
    v16bf a = make_fragA(a0);
    const unsigned short* bp = winp + ((size_t)(nt * 2 + kk) * 32 + lane) * 16;
    v16bf b = load_fragB(bp);
    acc = __builtin_amdgcn_wmma_f32_16x16x32_bf16(false, a, false, b, (short)0, acc,
                                                  false, false);
  }
  const int col = nt * 16 + li;
#pragma unroll
  for (int k = 0; k < 8; ++k) {
    int row = mt * 16 + k + lh * 8;
    pi[(size_t)row * NG6 + col] = f2bf_bits(acc[k]);
  }
}

// ---------------- persistent LSTM recurrence ----------------
// h and the step's pi slice are staged global->LDS with CDNA5 async copies
// (ASYNCcnt); h is double-buffered in 32x256 bf16 chunks overlapped with WMMA.

__device__ __forceinline__ void stage_chunk(const unsigned short* __restrict__ hcur,
                                            unsigned short (*h_lds)[32][256],
                                            int c, int buf, int tid) {
  if (tid < 256) {
#pragma unroll
    for (int q = 0; q < 4; ++q) {
      int i = q * 256 + tid;            // 1024 x 16B transfers = 32 rows x 512B
      int r = i >> 5;
      int seg = i & 31;
      const unsigned short* gp = hcur + (size_t)r * Hh + c * 256 + seg * 8;
      unsigned lo = (unsigned)(uintptr_t)(&h_lds[buf][r][seg * 8]);
      asm volatile("global_load_async_to_lds_b128 %0, %1, off"
                   :: "v"(lo), "v"(gp) : "memory");
    }
  }
}

// stage pi[b, t, g*2048 + j*16 .. +16] for all b,g : 32*6 segments of 32B
__device__ __forceinline__ void stage_pi(const unsigned short* __restrict__ pi,
                                         unsigned short (*pi_lds)[6][16],
                                         int j, int t, int tid) {
  if (tid < 192) {
#pragma unroll
    for (int q = 0; q < 2; ++q) {
      int i = q * 192 + tid;            // 384 x 16B transfers
      int b = i / 12;
      int rem = i - b * 12;
      int g = rem >> 1;
      int half = rem & 1;
      const unsigned short* gp =
          pi + ((size_t)b * Tt + t) * NG6 + g * Hh + j * 16 + half * 8;
      unsigned lo = (unsigned)(uintptr_t)(&pi_lds[b][g][half * 8]);
      asm volatile("global_load_async_to_lds_b128 %0, %1, off"
                   :: "v"(lo), "v"(gp) : "memory");
    }
  }
}

__global__ void __launch_bounds__(320, 1) k_lstm(
    const unsigned short* __restrict__ wsp,   // packed W_state bf16
    const unsigned short* __restrict__ pi,    // (B*T, 12288) bf16
    const float* __restrict__ b_state,        // 10240
    const float* __restrict__ mask,           // (B,T)
    unsigned short* __restrict__ hbuf,        // 2 x 32 x 2048 bf16 (double buffer)
    float* __restrict__ pooled,               // 32 x 2048 (masked sums)
    unsigned* __restrict__ bar) {
  __shared__ __align__(16) unsigned short h_lds[2][32][256];  // 32 KB
  __shared__ __align__(16) unsigned short pi_lds[32][6][16];  // 6 KB
  __shared__ float ps_lds[5][32][16];
  __shared__ float c_lds[32][16];
  __shared__ float pool_lds[32][16];
  __shared__ float bst_lds[5][16];

  const int j = blockIdx.x;            // hidden block 0..127 (16 cols each)
  const int tid = threadIdx.x;
  const int lane = tid & 31;
  const int wave = tid >> 5;           // 0..9
  const int g = wave >> 1;             // gate 0..4
  const int m = wave & 1;              // M-tile 0..1
  const int li = lane & 15, lh = lane >> 4;

  for (int idx = tid; idx < 512; idx += 320) {
    ((float*)c_lds)[idx] = 0.0f;
    ((float*)pool_lds)[idx] = 0.0f;
  }
  if (tid < 80) bst_lds[tid / 16][tid & 15] = b_state[(tid / 16) * Hh + j * 16 + (tid & 15)];
  __syncthreads();

  const unsigned short* wbase = wsp + (size_t)(g * 128 + j) * 64 * 32 * 16;
  const int row = m * 16 + li;

  for (int t = 0; t < Tt; ++t) {
    const unsigned short* hcur = hbuf + (size_t)(t & 1) * Bb * Hh;
    unsigned short* hnext = hbuf + (size_t)((t + 1) & 1) * Bb * Hh;

    // prologue: stage this step's pi slice + h chunk 0 (waited on before compute)
    stage_pi(pi, pi_lds, j, t, tid);
    stage_chunk(hcur, h_lds, 0, 0, tid);
    asm volatile("s_wait_asynccnt 0" ::: "memory");
    __syncthreads();

    // ps tile: 16x16 f32 = h(16x2048) @ W_state(2048x16); 8 chunks x 8 WMMA
    v8f acc = {};
    for (int c = 0; c < 8; ++c) {
      const int buf = c & 1;
      if (c < 7) stage_chunk(hcur, h_lds, c + 1, buf ^ 1, tid);
#pragma unroll
      for (int k8 = 0; k8 < 8; ++k8) {
        const unsigned short* a0 = &h_lds[buf][row][k8 * 32 + lh * 8];
        v16bf a = make_fragA(a0);                       // ds_load_b128 x2
        const int kk = c * 8 + k8;
        const unsigned short* bp = wbase + ((size_t)kk * 32 + lane) * 16;
        v16bf b = load_fragB(bp);                       // global b128 x2 (L2-resident)
        acc = __builtin_amdgcn_wmma_f32_16x16x32_bf16(false, a, false, b, (short)0, acc,
                                                      false, false);
      }
      asm volatile("s_wait_asynccnt 0" ::: "memory");
      __syncthreads();
    }
#pragma unroll
    for (int k = 0; k < 8; ++k)
      ps_lds[g][m * 16 + k + lh * 8][li] = acc[k];
    __syncthreads();

    // elementwise gates for the 32x16 block owned by this workgroup
    for (int idx = tid; idx < 512; idx += 320) {
      int b  = idx >> 4;
      int cl = idx & 15;
      int col = j * 16 + cl;
      float p0 = bf_bits2f(pi_lds[b][0][cl]);
      float p1 = bf_bits2f(pi_lds[b][1][cl]);
      float p2 = bf_bits2f(pi_lds[b][2][cl]);
      float p3 = bf_bits2f(pi_lds[b][3][cl]);
      float p4 = bf_bits2f(pi_lds[b][4][cl]);
      float p5 = bf_bits2f(pi_lds[b][5][cl]);
      float s0 = ps_lds[0][b][cl] + bst_lds[0][cl];
      float s1 = ps_lds[1][b][cl] + bst_lds[1][cl];
      float s2 = ps_lds[2][b][cl] + bst_lds[2][cl];
      float s3 = ps_lds[3][b][cl] + bst_lds[3][cl];
      float s4 = ps_lds[4][b][cl] + bst_lds[4][cl];
      float ig = sigmoidf(p0 + s0);
      float fg = sigmoidf(p1 + s1);
      float mi = tanhf(p2 + s2);
      float og = sigmoidf(p3 + s3);
      float hg = sigmoidf(p4 + s4);
      float cc = c_lds[b][cl];
      float cn = ig * mi + fg * cc;
      float o  = og * tanhf(cn);
      o = hg * o + (1.0f - hg) * p5;
      c_lds[b][cl] = cn;
      hnext[(size_t)b * Hh + col] = f2bf_bits(o);
      pool_lds[b][cl] += o * mask[b * Tt + t];
    }
    grid_barrier(bar, bar + 1, gridDim.x);
  }

  for (int idx = tid; idx < 512; idx += 320) {
    int b = idx >> 4, cl = idx & 15;
    pooled[(size_t)b * Hh + j * 16 + cl] = pool_lds[b][cl];
  }
}

// ---------------- classification head (single block) ----------------

__global__ void __launch_bounds__(1024, 1) k_head(
    const float* __restrict__ pooled, const float* __restrict__ mask,
    const float* __restrict__ g1, const float* __restrict__ be1,
    const float* __restrict__ W2, const float* __restrict__ b2,
    const float* __restrict__ g2, const float* __restrict__ be2,
    const float* __restrict__ W3, const float* __restrict__ b3,
    const int* __restrict__ target, float* __restrict__ xb, float* __restrict__ out) {
  __shared__ float msum[32];
  __shared__ float ob[32 * 100];
  __shared__ float logits[32 * 10];
  __shared__ float ce_s[32];
  __shared__ float cor_s[32];
  int tid = threadIdx.x;

  if (tid < 32) {
    float s = 0.0f;
    for (int t = 0; t < Tt; ++t) s += mask[tid * Tt + t];
    msum[tid] = s;
  }
  __syncthreads();

  // BN1 over batch axis per column, write normalized activations
  for (int col = tid; col < Hh; col += 1024) {
    float p[32]; float s1 = 0.0f, s2 = 0.0f;
    for (int b = 0; b < 32; ++b) {
      float v = pooled[b * Hh + col] / msum[b];
      p[b] = v; s1 += v; s2 += v * v;
    }
    float mn = s1 * (1.0f / 32.0f);
    float var = s2 * (1.0f / 32.0f) - mn * mn;
    float inv = rsqrtf(var + 1e-5f) * g1[col];
    float bt = be1[col];
    for (int b = 0; b < 32; ++b) xb[b * Hh + col] = (p[b] - mn) * inv + bt;
  }
  __syncthreads();

  // xb(32x2048) @ W2(2048x100) + b2
  for (int idx = tid; idx < 3200; idx += 1024) {
    int b = idx / 100, n = idx - b * 100;
    float s = b2[n];
    for (int k = 0; k < Hh; ++k) s += xb[b * Hh + k] * W2[k * 100 + n];
    ob[idx] = s;
  }
  __syncthreads();

  // BN2
  if (tid < 100) {
    float s1 = 0.0f, s2 = 0.0f;
    for (int b = 0; b < 32; ++b) { float v = ob[b * 100 + tid]; s1 += v; s2 += v * v; }
    float mn = s1 * (1.0f / 32.0f);
    float var = s2 * (1.0f / 32.0f) - mn * mn;
    float inv = rsqrtf(var + 1e-5f) * g2[tid];
    float bt = be2[tid];
    for (int b = 0; b < 32; ++b) ob[b * 100 + tid] = (ob[b * 100 + tid] - mn) * inv + bt;
  }
  __syncthreads();

  // logits = bn2 @ W3(100x10) + b3
  if (tid < 320) {
    int b = tid / 10, o = tid - b * 10;
    float s = b3[o];
    for (int k = 0; k < 100; ++k) s += ob[b * 100 + k] * W3[k * 10 + o];
    logits[tid] = s;
  }
  __syncthreads();

  // log-softmax, CE, accuracy
  if (tid < 32) {
    float mx = -1e30f; int arg = 0;
    for (int o = 0; o < 10; ++o) {
      float v = logits[tid * 10 + o];
      if (v > mx) { mx = v; arg = o; }
    }
    float se = 0.0f;
    for (int o = 0; o < 10; ++o) se += __expf(logits[tid * 10 + o] - mx);
    float lse = __logf(se) + mx;
    int tg = target[tid];
    ce_s[tid]  = -(logits[tid * 10 + tg] - lse);
    cor_s[tid] = (arg == tg) ? 1.0f : 0.0f;
  }
  __syncthreads();
  if (tid == 0) {
    float ce = 0.0f, ac = 0.0f;
    for (int b = 0; b < 32; ++b) { ce += ce_s[b]; ac += cor_s[b]; }
    out[0] = ac * (1.0f / 32.0f);   // acc
    out[1] = ce * (1.0f / 32.0f);   // ce_loss
  }
}

// ---------------- launch ----------------

extern "C" void kernel_launch(void* const* d_in, const int* in_sizes, int n_in,
                              void* d_out, int out_size, void* d_ws, size_t ws_size,
                              hipStream_t stream) {
  const float* src     = (const float*)d_in[0];
  const float* mask    = (const float*)d_in[1];
  const int*   target  = (const int*)d_in[2];
  const float* W_in    = (const float*)d_in[3];
  const float* W_state = (const float*)d_in[4];
  const float* b_state = (const float*)d_in[5];
  const float* g1      = (const float*)d_in[6];
  const float* be1     = (const float*)d_in[7];
  const float* W2      = (const float*)d_in[8];
  const float* b2      = (const float*)d_in[9];
  const float* g2      = (const float*)d_in[10];
  const float* be2     = (const float*)d_in[11];
  const float* W3      = (const float*)d_in[12];
  const float* b3      = (const float*)d_in[13];
  float* out = (float*)d_out;

  char* ws = (char*)d_ws;
  size_t off = 0;
  auto take = [&](size_t bytes) -> char* {
    char* p = ws + off;
    off = (off + bytes + 255) & ~(size_t)255;
    return p;
  };
  unsigned short* wsp    = (unsigned short*)take((size_t)2048 * 10240 * 2); // 41.9 MB
  unsigned short* winp   = (unsigned short*)take((size_t)64 * 12288 * 2);   // 1.6 MB
  unsigned short* srcp   = (unsigned short*)take((size_t)16384 * 64 * 2);   // 2.1 MB
  unsigned short* hbuf   = (unsigned short*)take((size_t)2 * 32 * 2048 * 2);// 256 KB
  unsigned*       bar    = (unsigned*)take(256);
  float*          pooled = (float*)take((size_t)32 * 2048 * 4);
  float*          xb     = (float*)take((size_t)32 * 2048 * 4);
  unsigned short* pi     = (unsigned short*)take((size_t)16384 * 12288 * 2);// 402 MB

  k_zero<<<(65536 + 255) / 256, 256, 0, stream>>>((unsigned*)hbuf, bar);
  k_pack_ws<<<(20971520 + 255) / 256, 256, 0, stream>>>(W_state, wsp);
  k_pack_win<<<(786432 + 255) / 256, 256, 0, stream>>>(W_in, winp);
  k_pack_src<<<(1048576 + 255) / 256, 256, 0, stream>>>(src, srcp);
  k_gemm_pi<<<98304, 256, 0, stream>>>(srcp, winp, pi);
  k_lstm<<<128, 320, 0, stream>>>(wsp, pi, b_state, mask, hbuf, pooled, bar);
  k_head<<<1, 1024, 0, stream>>>(pooled, mask, g1, be1, W2, b2, g2, be2, W3, b3,
                                 target, xb, out);
}
